// Model_51702816309366
// MI455X (gfx1250) — matrix operand order repaired and checked by reading, not compiled, not run on hardware
//
#include <hip/hip_runtime.h>

// ---------------------------------------------------------------------------
// GNN (bipartite half-conv) for MI455X / gfx1250.
//   - fp16 storage of all 64-wide node features (halves L2/HBM gather traffic;
//     every feature table is <=12.8MB so the whole working set sits in 192MB L2)
//   - all matmuls via V_WMMA_F32_16X16X32_F16 (f32 accumulate)
//   - edge MLP fully fused: gather -> [160->64] relu -> [64->64] relu ->
//     f32 atomic scatter-add (never materializes the E x 64 intermediate)
//   - B-fragments for all 4 N-tiles are loaded before the WMMA chain so the
//     scheduler can overlap ds_load latency with XDL issue (partial dscnt
//     waits instead of s_wait_dscnt 0 before every WMMA)
// ---------------------------------------------------------------------------

typedef _Float16 h16;
typedef __attribute__((ext_vector_type(8)))  _Float16 v8h;
typedef __attribute__((ext_vector_type(16))) _Float16 v16h;
typedef __attribute__((ext_vector_type(8)))  float    v8f;

#define WPB 4              // waves per block
#define BLOCK_THREADS 128  // 4 x wave32

static __device__ __forceinline__ v8f vzero8() {
  v8f z = {0.f, 0.f, 0.f, 0.f, 0.f, 0.f, 0.f, 0.f};
  return z;
}

// One K-chunk (32) of D(16x64) += A(16xK) * B(Kx64), f16 in / f32 acc.
// A-fragment: lanes 0-15 hold K 0..15,
// lanes 16-31 hold K 16..31 for row M = lane&15.
// B-fragment: mirrored striping, lane = column N (&15), lanes 0-15 K 0..15,
// lanes 16-31 K 16..31 of the chunk. B is stored column-major ([n][k]) in LDS
// so each lane reads 16 contiguous halves (32B, ds_load_b128 x2).
// All 4 B fragments are fetched before the WMMA chain (latency hiding).
static __device__ __forceinline__ void wmma_k32(const h16* As, int strideA, int kc,
                                                const h16* Bt, int Kp,
                                                v8f acc[4], int lane) {
  const int m   = lane & 15;
  const int kbA = (lane & 16) ? 16 : 0;
  const v8h a0 = *(const v8h*)(As + m * strideA + kc + kbA);
  const v8h a1 = *(const v8h*)(As + m * strideA + kc + 8 + kbA);
  const v16h a = __builtin_shufflevector(a0, a1, 0, 1, 2, 3, 4, 5, 6, 7,
                                         8, 9, 10, 11, 12, 13, 14, 15);
  const int n   = lane & 15;
  const int kbB = (lane & 16) ? 16 : 0;
  v16h b[4];
#pragma unroll
  for (int nt = 0; nt < 4; ++nt)
    b[nt] = *(const v16h*)(Bt + (nt * 16 + n) * Kp + kc + kbB);
#pragma unroll
  for (int nt = 0; nt < 4; ++nt)
    acc[nt] = __builtin_amdgcn_wmma_f32_16x16x32_f16(
        false, a, false, b[nt], (short)0, acc[nt], false, false);
}

// Full D(16x64) = A(16xK) * B(Kx64); K multiple of 32; A row-major stride K,
// B column-major [64][K] in LDS.
static __device__ __forceinline__ void gemm16xKx64(const h16* As, int K,
                                                   const h16* Bt, v8f acc[4],
                                                   int lane) {
  for (int kc = 0; kc < K; kc += 32) wmma_k32(As, K, kc, Bt, K, acc, lane);
}

// acc (C/D layout: vgpr r -> row M = r + (lane>=16)*8, col N = nt*16 + lane&15)
// + bias, relu, store f16 to 16x64 row-major LDS tile.
static __device__ __forceinline__ void acc_store_h(const v8f acc[4],
                                                   const float* __restrict__ bias,
                                                   int bn, h16* H, int lane) {
  const int mB = (lane & 16) ? 8 : 0;
  const int nl = lane & 15;
#pragma unroll
  for (int nt = 0; nt < 4; ++nt) {
    const int n = nt * 16 + nl;
    const float bb = (n < bn) ? bias[n] : 0.f;
#pragma unroll
    for (int r = 0; r < 8; ++r) {
      float x = acc[nt][r] + bb;
      x = fmaxf(x, 0.f);
      H[(mB + r) * 64 + n] = (h16)x;
    }
  }
}

// fp32 W[K0][N0] -> f16, transposed + zero-padded to Wt[Np(=64)][Kp].
__global__ void prep_weight_kernel(const float* __restrict__ W, int K0, int N0,
                                   h16* __restrict__ Wt, int Kp) {
  const int i = blockIdx.x * blockDim.x + threadIdx.x;
  if (i >= Kp * 64) return;
  const int n = i / Kp;
  const int k = i - n * Kp;
  const float x = (k < K0 && n < N0) ? W[k * N0 + n] : 0.f;
  Wt[n * Kp + k] = (h16)x;
}

// Node embedding: X (N x D fp32) -> [D(pad32)->64] relu -> [64->64] relu -> f16 out.
__global__ __launch_bounds__(BLOCK_THREADS) void embed_kernel(
    const float* __restrict__ X, int D, int N,
    const h16* __restrict__ Wt1, const float* __restrict__ b1,
    const h16* __restrict__ Wt2, const float* __restrict__ b2,
    h16* __restrict__ out) {
  __shared__ alignas(32) h16 W1s[64 * 32];
  __shared__ alignas(32) h16 W2s[64 * 64];
  __shared__ alignas(32) h16 As[WPB][16 * 32];
  __shared__ alignas(32) h16 Hs[WPB][16 * 64];

  const int tid = threadIdx.x;
  for (int i = tid; i < 64 * 32; i += BLOCK_THREADS) W1s[i] = Wt1[i];
  for (int i = tid; i < 64 * 64; i += BLOCK_THREADS) W2s[i] = Wt2[i];

  const int wave = tid >> 5, lane = tid & 31;
  h16* A = As[wave];
  h16* H = Hs[wave];
  const long r0 = ((long)blockIdx.x * WPB + wave) * 16;

  if (lane < 16) {
    const long row = r0 + lane;
    const bool ok = row < (long)N;
    for (int t = 0; t < 32; ++t) {
      const float x = (ok && t < D) ? X[row * D + t] : 0.f;
      A[lane * 32 + t] = (h16)x;
    }
  }
  __syncthreads();

  v8f acc[4] = {vzero8(), vzero8(), vzero8(), vzero8()};
  gemm16xKx64(A, 32, W1s, acc, lane);
  acc_store_h(acc, b1, 64, H, lane);
  __syncthreads();

  v8f acc2[4] = {vzero8(), vzero8(), vzero8(), vzero8()};
  gemm16xKx64(H, 64, W2s, acc2, lane);
  acc_store_h(acc2, b2, 64, H, lane);
  __syncthreads();

  if (lane < 16) {
    const long row = r0 + lane;
    if (row < (long)N) {
#pragma unroll
      for (int t = 0; t < 8; ++t)
        *(uint4*)(out + row * 64 + t * 8) = *(const uint4*)(H + lane * 64 + t * 8);
    }
  }
}

// Fused half-conv "g" MLP + scatter: per 16-edge tile gather
// [u[uIdx] | op[opIdx] | e_val | 0-pad] (16x160), [160->64] relu -> [64->64]
// relu -> atomicAdd into agg[uIdx].
__global__ __launch_bounds__(BLOCK_THREADS) void edge_conv_kernel(
    const h16* __restrict__ uF, const h16* __restrict__ opF,
    const int* __restrict__ uIdxG, const int* __restrict__ opIdxG,
    const float* __restrict__ eVal,
    const h16* __restrict__ Wt1, const float* __restrict__ b1,
    const h16* __restrict__ Wt2, const float* __restrict__ b2,
    float* __restrict__ agg, int E) {
  __shared__ alignas(32) h16 W1s[64 * 160];
  __shared__ alignas(32) h16 W2s[64 * 64];
  __shared__ alignas(32) h16 As[WPB][16 * 160];
  __shared__ alignas(32) h16 Hs[WPB][16 * 64];
  __shared__ int Eu[WPB][16];

  const int tid = threadIdx.x;
  for (int i = tid; i < 64 * 160; i += BLOCK_THREADS) W1s[i] = Wt1[i];
  for (int i = tid; i < 64 * 64; i += BLOCK_THREADS) W2s[i] = Wt2[i];

  const int wave = tid >> 5, lane = tid & 31;
  h16* A = As[wave];
  h16* H = Hs[wave];
  int* eu = Eu[wave];
  const long e0 = ((long)blockIdx.x * WPB + wave) * 16;

  {  // gather: lanes 0-15 -> u rows into cols 0..63 (+meta); 16-31 -> op rows into 64..127
    const int m = lane & 15;
    const long e = e0 + m;
    const bool ok = e < (long)E;
    const long ec = ok ? e : 0;
    const h16* src;
    h16* dst;
    if (lane < 16) {
      const int ui = uIdxG[ec];
      eu[m] = ok ? ui : -1;
      A[m * 160 + 128] = (h16)(ok ? eVal[ec] : 0.f);
#pragma unroll
      for (int t = 129; t < 160; ++t) A[m * 160 + t] = (h16)0.f;
      src = uF + (long)ui * 64;
      dst = A + m * 160;
    } else {
      const int oi = opIdxG[ec];
      src = opF + (long)oi * 64;
      dst = A + m * 160 + 64;
    }
    if (ok) {
#pragma unroll
      for (int t = 0; t < 8; ++t)
        *(uint4*)(dst + t * 8) = *(const uint4*)(src + t * 8);
    } else {
      const uint4 z = {0u, 0u, 0u, 0u};
#pragma unroll
      for (int t = 0; t < 8; ++t) *(uint4*)(dst + t * 8) = z;
    }
  }
  __syncthreads();

  v8f acc[4] = {vzero8(), vzero8(), vzero8(), vzero8()};
  gemm16xKx64(A, 160, W1s, acc, lane);
  acc_store_h(acc, b1, 64, H, lane);
  __syncthreads();

  v8f acc2[4] = {vzero8(), vzero8(), vzero8(), vzero8()};
  gemm16xKx64(H, 64, W2s, acc2, lane);

  // bias + relu + segment-sum scatter (f32 global atomics; L2-resident)
  const int mB = (lane & 16) ? 8 : 0;
  const int nl = lane & 15;
#pragma unroll
  for (int nt = 0; nt < 4; ++nt) {
    const int n = nt * 16 + nl;
    const float bb = b2[n];
#pragma unroll
    for (int r = 0; r < 8; ++r) {
      const int row = eu[mB + r];
      if (row >= 0) {
        const float x = fmaxf(acc2[nt][r] + bb, 0.f);
        atomicAdd(agg + (size_t)row * 64 + n, x);
      }
    }
  }
}

// Node update "f": concat(u f16, agg f32->f16) (16x128) -> [128->64] relu ->
// [64->64] relu -> f16 out.
__global__ __launch_bounds__(BLOCK_THREADS) void update_kernel(
    const h16* __restrict__ U, const float* __restrict__ AGG, int N,
    const h16* __restrict__ Wt1, const float* __restrict__ b1,
    const h16* __restrict__ Wt2, const float* __restrict__ b2,
    h16* __restrict__ out) {
  __shared__ alignas(32) h16 W1s[64 * 128];
  __shared__ alignas(32) h16 W2s[64 * 64];
  __shared__ alignas(32) h16 As[WPB][16 * 128];
  __shared__ alignas(32) h16 Hs[WPB][16 * 64];

  const int tid = threadIdx.x;
  for (int i = tid; i < 64 * 128; i += BLOCK_THREADS) W1s[i] = Wt1[i];
  for (int i = tid; i < 64 * 64; i += BLOCK_THREADS) W2s[i] = Wt2[i];

  const int wave = tid >> 5, lane = tid & 31;
  h16* A = As[wave];
  h16* H = Hs[wave];
  const long r0 = ((long)blockIdx.x * WPB + wave) * 16;

  {
    const int m = lane & 15;
    const long row = r0 + m;
    const bool ok = row < (long)N;
    if (lane < 16) {
      if (ok) {
#pragma unroll
        for (int t = 0; t < 8; ++t)
          *(uint4*)(A + m * 128 + t * 8) = *(const uint4*)(U + row * 64 + t * 8);
      } else {
        const uint4 z = {0u, 0u, 0u, 0u};
#pragma unroll
        for (int t = 0; t < 8; ++t) *(uint4*)(A + m * 128 + t * 8) = z;
      }
    } else {
      if (ok) {
        const float4* ar = (const float4*)(AGG + row * 64);
#pragma unroll
        for (int t = 0; t < 16; ++t) {
          const float4 f = ar[t];
          A[m * 128 + 64 + 4 * t + 0] = (h16)f.x;
          A[m * 128 + 64 + 4 * t + 1] = (h16)f.y;
          A[m * 128 + 64 + 4 * t + 2] = (h16)f.z;
          A[m * 128 + 64 + 4 * t + 3] = (h16)f.w;
        }
      } else {
        for (int t = 0; t < 64; ++t) A[m * 128 + 64 + t] = (h16)0.f;
      }
    }
  }
  __syncthreads();

  v8f acc[4] = {vzero8(), vzero8(), vzero8(), vzero8()};
  gemm16xKx64(A, 128, W1s, acc, lane);
  acc_store_h(acc, b1, 64, H, lane);
  __syncthreads();

  v8f acc2[4] = {vzero8(), vzero8(), vzero8(), vzero8()};
  gemm16xKx64(H, 64, W2s, acc2, lane);
  acc_store_h(acc2, b2, 64, H, lane);
  __syncthreads();

  if (lane < 16) {
    const long row = r0 + lane;
    if (row < (long)N) {
#pragma unroll
      for (int t = 0; t < 8; ++t)
        *(uint4*)(out + row * 64 + t * 8) = *(const uint4*)(H + lane * 64 + t * 8);
    }
  }
}

// Tail: v2 (16x64 f16) -> [64->64] relu -> [64->8 (padded to 64)] -> sigmoid f32.
__global__ __launch_bounds__(BLOCK_THREADS) void tail_kernel(
    const h16* __restrict__ V2, int N,
    const h16* __restrict__ Wt1, const float* __restrict__ b1,
    const h16* __restrict__ Wt2, const float* __restrict__ b2,
    float* __restrict__ out) {
  __shared__ alignas(32) h16 W1s[64 * 64];
  __shared__ alignas(32) h16 W2s[64 * 64];  // cols >= 8 are zero-padded
  __shared__ alignas(32) h16 As[WPB][16 * 64];
  __shared__ alignas(32) h16 Hs[WPB][16 * 64];

  const int tid = threadIdx.x;
  for (int i = tid; i < 64 * 64; i += BLOCK_THREADS) {
    W1s[i] = Wt1[i];
    W2s[i] = Wt2[i];
  }

  const int wave = tid >> 5, lane = tid & 31;
  h16* A = As[wave];
  h16* H = Hs[wave];
  const long r0 = ((long)blockIdx.x * WPB + wave) * 16;

  if (lane < 16) {
    const long row = r0 + lane;
    if (row < (long)N) {
#pragma unroll
      for (int t = 0; t < 8; ++t)
        *(uint4*)(A + lane * 64 + t * 8) = *(const uint4*)(V2 + row * 64 + t * 8);
    } else {
      const uint4 z = {0u, 0u, 0u, 0u};
#pragma unroll
      for (int t = 0; t < 8; ++t) *(uint4*)(A + lane * 64 + t * 8) = z;
    }
  }
  __syncthreads();

  v8f acc[4] = {vzero8(), vzero8(), vzero8(), vzero8()};
  gemm16xKx64(A, 64, W1s, acc, lane);
  acc_store_h(acc, b1, 64, H, lane);
  __syncthreads();

  v8f acc2[4] = {vzero8(), vzero8(), vzero8(), vzero8()};
  gemm16xKx64(H, 64, W2s, acc2, lane);

  const int mB = (lane & 16) ? 8 : 0;
  const int nl = lane & 15;
  if (nl < 8) {
#pragma unroll
    for (int r = 0; r < 8; ++r) {
      const long row = r0 + mB + r;
      if (row < (long)N) {
        const float x = acc2[0][r] + b2[nl];
        out[row * 8 + nl] = 1.f / (1.f + __expf(-x));
      }
    }
  }
}

// ---------------------------------------------------------------------------
// d_in layout (setup_inputs dict order; param lists flatten (W,b) per layer):
//  0 v(NV*19) 1 c(NC*5) 2 e_val(E) 3 e_row(E,int) 4 e_col(E,int)
//  5/6/7/8    emb_v  W0(19x64) b0 W1(64x64) b1
//  9/10/11/12 emb_c  W0(5x64)  b0 W1        b1
// 13..16 gc (129x64 / 64x64)   17..20 fc (128x64 / 64x64)
// 21..24 gv (129x64 / 64x64)   25..28 fv (128x64 / 64x64)
// 29..32 tail (64x64 / 64x8)
// ---------------------------------------------------------------------------
extern "C" void kernel_launch(void* const* d_in, const int* in_sizes, int n_in,
                              void* d_out, int out_size, void* d_ws, size_t ws_size,
                              hipStream_t stream) {
  (void)n_in; (void)out_size; (void)ws_size;

  const int NV = in_sizes[0] / 19;
  const int NC = in_sizes[1] / 5;
  const int E  = in_sizes[2];

  const float* v_in  = (const float*)d_in[0];
  const float* c_in  = (const float*)d_in[1];
  const float* e_val = (const float*)d_in[2];
  const int*   e_row = (const int*)d_in[3];
  const int*   e_col = (const int*)d_in[4];

  char* wsb = (char*)d_ws;
  size_t off = 0;
  auto balloc = [&](size_t bytes) -> void* {
    void* p = wsb + off;
    off += (bytes + 255) & ~(size_t)255;
    return p;
  };

  h16*   v1   = (h16*)balloc((size_t)NV * 64 * sizeof(h16));
  h16*   c1   = (h16*)balloc((size_t)NC * 64 * sizeof(h16));
  h16*   c2   = (h16*)balloc((size_t)NC * 64 * sizeof(h16));
  h16*   v2   = (h16*)balloc((size_t)NV * 64 * sizeof(h16));
  float* aggC = (float*)balloc((size_t)NC * 64 * sizeof(float));
  float* aggV = (float*)balloc((size_t)NV * 64 * sizeof(float));

  // f16 transposed+padded weights
  auto prepW = [&](int idx, int K0, int N0, int Kp) -> h16* {
    h16* p = (h16*)balloc((size_t)Kp * 64 * sizeof(h16));
    const int total = Kp * 64;
    prep_weight_kernel<<<(total + 255) / 256, 256, 0, stream>>>(
        (const float*)d_in[idx], K0, N0, p, Kp);
    return p;
  };
  h16* Wev1 = prepW(5, 19, 64, 32);
  h16* Wev2 = prepW(7, 64, 64, 64);
  h16* Wec1 = prepW(9, 5, 64, 32);
  h16* Wec2 = prepW(11, 64, 64, 64);
  h16* Wgc1 = prepW(13, 129, 64, 160);
  h16* Wgc2 = prepW(15, 64, 64, 64);
  h16* Wfc1 = prepW(17, 128, 64, 128);
  h16* Wfc2 = prepW(19, 64, 64, 64);
  h16* Wgv1 = prepW(21, 129, 64, 160);
  h16* Wgv2 = prepW(23, 64, 64, 64);
  h16* Wfv1 = prepW(25, 128, 64, 128);
  h16* Wfv2 = prepW(27, 64, 64, 64);
  h16* Wt1w = prepW(29, 64, 64, 64);
  h16* Wt2w = prepW(31, 64, 8, 64);  // N padded 8 -> 64 (zeros)

  const float* bev1 = (const float*)d_in[6];
  const float* bev2 = (const float*)d_in[8];
  const float* bec1 = (const float*)d_in[10];
  const float* bec2 = (const float*)d_in[12];
  const float* bgc1 = (const float*)d_in[14];
  const float* bgc2 = (const float*)d_in[16];
  const float* bfc1 = (const float*)d_in[18];
  const float* bfc2 = (const float*)d_in[20];
  const float* bgv1 = (const float*)d_in[22];
  const float* bgv2 = (const float*)d_in[24];
  const float* bfv1 = (const float*)d_in[26];
  const float* bfv2 = (const float*)d_in[28];
  const float* bt1  = (const float*)d_in[30];
  const float* bt2  = (const float*)d_in[32];

  // zero the segment-sum accumulators every call (atomics accumulate)
  hipMemsetAsync(aggC, 0, (size_t)NC * 64 * sizeof(float), stream);
  hipMemsetAsync(aggV, 0, (size_t)NV * 64 * sizeof(float), stream);

  const int nodeBlocksV = (NV + 16 * WPB - 1) / (16 * WPB);
  const int nodeBlocksC = (NC + 16 * WPB - 1) / (16 * WPB);
  const int edgeBlocks  = (E + 16 * WPB - 1) / (16 * WPB);

  // v1 = emb_v(v), c1 = emb_c(c)
  embed_kernel<<<nodeBlocksV, BLOCK_THREADS, 0, stream>>>(
      v_in, 19, NV, Wev1, bev1, Wev2, bev2, v1);
  embed_kernel<<<nodeBlocksC, BLOCK_THREADS, 0, stream>>>(
      c_in, 5, NC, Wec1, bec1, Wec2, bec2, c1);

  // c-side conv: g over edges (u=c1[e_row], op=v1[e_col]) -> aggC; c2 = fc(c1, aggC)
  edge_conv_kernel<<<edgeBlocks, BLOCK_THREADS, 0, stream>>>(
      c1, v1, e_row, e_col, e_val, Wgc1, bgc1, Wgc2, bgc2, aggC, E);
  update_kernel<<<nodeBlocksC, BLOCK_THREADS, 0, stream>>>(
      c1, aggC, NC, Wfc1, bfc1, Wfc2, bfc2, c2);

  // v-side conv: g over edges (u=v1[e_col], op=c2[e_row]) -> aggV; v2 = fv(v1, aggV)
  edge_conv_kernel<<<edgeBlocks, BLOCK_THREADS, 0, stream>>>(
      v1, c2, e_col, e_row, e_val, Wgv1, bgv1, Wgv2, bgv2, aggV, E);
  update_kernel<<<nodeBlocksV, BLOCK_THREADS, 0, stream>>>(
      v1, aggV, NV, Wfv1, bfv1, Wfv2, bfv2, v2);

  // tail + sigmoid -> d_out (NV x 8 f32)
  tail_kernel<<<nodeBlocksV, BLOCK_THREADS, 0, stream>>>(
      v2, NV, Wt1w, bt1, Wt2w, bt2, (float*)d_out);
}
